// RGCNLayer_48086453846016
// MI455X (gfx1250) — matrix operand order repaired and checked by reading, not compiled, not run on hardware
//
#include <hip/hip_runtime.h>

// ---------------------------------------------------------------------------
// RGCN layer for MI455X (gfx1250, wave32, WMMA 16x16x32 bf16, fp32 accum)
//   out = LN( x @ W_self  +  scatter_add(dst, (x[src]) @ W_rel) )
// Edge kernel stages gathered A rows in LDS (shared by all 8 waves) and
// amortizes each weight-fragment load over 4 row-tiles (64 edges/block).
// ---------------------------------------------------------------------------

#define N_NODES   100000
#define DIM       256
#define N_REL     8
#define E_PER_REL 160000
#define LN_EPS    1e-5f

#define EDGE_TILE 64                  // edges per block (4 row-tiles of 16)
#define LDS_STRIDE 264                // halves per LDS row (padded vs 256)

typedef __attribute__((ext_vector_type(16))) __bf16 v16bf;
typedef __attribute__((ext_vector_type(8)))  float  v8f;
typedef __attribute__((ext_vector_type(8)))  unsigned int u32x8;

// float -> bf16 with round-to-nearest-even
__device__ __forceinline__ unsigned short f2bf(float f) {
    unsigned int u = __builtin_bit_cast(unsigned int, f);
    u += 0x7FFFu + ((u >> 16) & 1u);
    return (unsigned short)(u >> 16);
}

// Build a v16bf WMMA fragment from two 16-byte chunks (global or LDS).
__device__ __forceinline__ v16bf load_frag(const unsigned short* c0,
                                           const unsigned short* c1) {
    uint4 lo = *reinterpret_cast<const uint4*>(c0);
    uint4 hi = *reinterpret_cast<const uint4*>(c1);
    u32x8 u = {lo.x, lo.y, lo.z, lo.w, hi.x, hi.y, hi.z, hi.w};
    return __builtin_bit_cast(v16bf, u);
}

// ------------------------- conversion kernels ------------------------------

__global__ void __launch_bounds__(256) k_convert_x(const float* __restrict__ x,
                                                   unsigned short* __restrict__ xb,
                                                   int n4) {
    int i = blockIdx.x * 256 + threadIdx.x;
    if (i < n4) {
        float4 v = reinterpret_cast<const float4*>(x)[i];
        ushort4 o;
        o.x = f2bf(v.x); o.y = f2bf(v.y); o.z = f2bf(v.z); o.w = f2bf(v.w);
        reinterpret_cast<ushort4*>(xb)[i] = o;
    }
}

// W_self [K,O] and rel_weight [R,K,O] -> transposed bf16 Wt[r][o][k]
__global__ void __launch_bounds__(256) k_convert_w(const float* __restrict__ Wself,
                                                   const float* __restrict__ Wrel,
                                                   unsigned short* __restrict__ wb) {
    int idx = blockIdx.x * 256 + threadIdx.x;            // < 9*256*256
    int rr  = idx >> 16;
    int rem = idx & 65535;
    int o   = rem >> 8;
    int k   = rem & 255;
    float v = (rr == 0) ? Wself[k * DIM + o]
                        : Wrel[(size_t)(rr - 1) * DIM * DIM + k * DIM + o];
    wb[idx] = f2bf(v);
}

// --------------------------- self-loop GEMM --------------------------------
__global__ void __launch_bounds__(256) k_self_gemm(const unsigned short* __restrict__ xb,
                                                   const unsigned short* __restrict__ wb,
                                                   float* __restrict__ out) {
    const int tile = blockIdx.x;                 // 0 .. 6249
    const int wave = threadIdx.x >> 5;
    const int lane = threadIdx.x & 31;
    const int g    = lane >> 4;
    const int rl   = lane & 15;

    const int row = tile * 16 + rl;
    const unsigned short* arow = xb + (size_t)row * DIM;

    v16bf afrag[8];
#pragma unroll
    for (int kk = 0; kk < 8; ++kk)
        afrag[kk] = load_frag(arow + kk * 32 + g * 8,
                              arow + kk * 32 + 16 + g * 8);

#pragma unroll
    for (int cset = 0; cset < 2; ++cset) {
        const int col0 = (wave + cset * 8) * 16;
        const int col  = col0 + rl;
        const unsigned short* brow = wb + (size_t)col * DIM;

        v8f acc = (v8f)0.0f;
#pragma unroll
        for (int kk = 0; kk < 8; ++kk) {
            v16bf b = load_frag(brow + kk * 32 + g * 8,
                                brow + kk * 32 + 16 + g * 8);
            acc = __builtin_amdgcn_wmma_f32_16x16x32_bf16(
                false, afrag[kk], false, b, (short)0, acc, false, false);
        }
#pragma unroll
        for (int j = 0; j < 8; ++j) {
            int m = g * 8 + j;
            out[(size_t)(tile * 16 + m) * DIM + col] = acc[j];
        }
    }
}

// ------------------------ relational gather-GEMM ---------------------------
// Block: 64 edges x 256 cols. A rows gathered once into LDS; each wave does
// 2 col-tiles x 4 row-tiles, reusing every B fragment 4x from registers.
__global__ void __launch_bounds__(256) k_edge_gemm(const unsigned short* __restrict__ xb,
                                                   const unsigned short* __restrict__ wb,
                                                   const int* __restrict__ src,
                                                   const int* __restrict__ dst,
                                                   float* __restrict__ out) {
    __shared__ unsigned short sA[EDGE_TILE * LDS_STRIDE];
    __shared__ int sdst[EDGE_TILE];

    const int bid  = blockIdx.x;                 // 0 .. 19999
    const int rel  = bid / (E_PER_REL / EDGE_TILE);
    const int tile = bid % (E_PER_REL / EDGE_TILE);
    const int tid  = threadIdx.x;
    const int ebase = rel * E_PER_REL + tile * EDGE_TILE;

    // --- cooperative gather: 64 rows x 512B -> LDS (128B per thread) ---
    {
        const int row  = tid >> 2;               // 0..63
        const int part = tid & 3;                // 0..3 (128B chunk)
        const int srow = src[ebase + row];
        const uint4* gp = reinterpret_cast<const uint4*>(xb + (size_t)srow * DIM) + part * 8;
        uint4* lp = reinterpret_cast<uint4*>(&sA[row * LDS_STRIDE + part * 64]);
#pragma unroll
        for (int j = 0; j < 8; ++j) lp[j] = gp[j];
    }
    if (tid < EDGE_TILE) sdst[tid] = dst[ebase + tid];
    __syncthreads();

    const int wave = tid >> 5;
    const int lane = tid & 31;
    const int g    = lane >> 4;
    const int rl   = lane & 15;

    const unsigned short* wt = wb + (size_t)(rel + 1) * DIM * DIM;

#pragma unroll
    for (int cset = 0; cset < 2; ++cset) {
        const int col0 = (wave + cset * 8) * 16;
        const int col  = col0 + rl;
        const unsigned short* brow = wt + (size_t)col * DIM;

        v8f acc[4];
#pragma unroll
        for (int rt = 0; rt < 4; ++rt) acc[rt] = (v8f)0.0f;

#pragma unroll
        for (int kk = 0; kk < 8; ++kk) {
            v16bf b = load_frag(brow + kk * 32 + g * 8,
                                brow + kk * 32 + 16 + g * 8);
#pragma unroll
            for (int rt = 0; rt < 4; ++rt) {
                const unsigned short* ar =
                    &sA[(rt * 16 + rl) * LDS_STRIDE + kk * 32 + g * 8];
                v16bf a = load_frag(ar, ar + 16);   // ds_load_b128 x2
                acc[rt] = __builtin_amdgcn_wmma_f32_16x16x32_bf16(
                    false, a, false, b, (short)0, acc[rt], false, false);
            }
        }

        // scatter-add: lane holds M = rt*16 + g*8 + j, N = col
#pragma unroll
        for (int rt = 0; rt < 4; ++rt) {
#pragma unroll
            for (int j = 0; j < 8; ++j) {
                int dr = sdst[rt * 16 + g * 8 + j];
                atomicAdd(&out[(size_t)dr * DIM + col], acc[rt][j]);
            }
        }
    }
}

// ------------------------------ LayerNorm ----------------------------------
__global__ void __launch_bounds__(256) k_layernorm(float* __restrict__ out,
                                                   const float* __restrict__ gamma,
                                                   const float* __restrict__ beta) {
    const int wave = threadIdx.x >> 5;
    const int lane = threadIdx.x & 31;
    const int row  = blockIdx.x * 8 + wave;      // grid = 12500

    float* p = out + (size_t)row * DIM + lane * 8;
    float4 v0 = reinterpret_cast<float4*>(p)[0];
    float4 v1 = reinterpret_cast<float4*>(p)[1];

    float s = v0.x + v0.y + v0.z + v0.w + v1.x + v1.y + v1.z + v1.w;
    float q = v0.x * v0.x + v0.y * v0.y + v0.z * v0.z + v0.w * v0.w +
              v1.x * v1.x + v1.y * v1.y + v1.z * v1.z + v1.w * v1.w;

#pragma unroll
    for (int m = 16; m >= 1; m >>= 1) {
        s += __shfl_xor(s, m, 32);
        q += __shfl_xor(q, m, 32);
    }

    const float mean = s * (1.0f / DIM);
    const float var  = q * (1.0f / DIM) - mean * mean;
    const float rstd = rsqrtf(var + LN_EPS);

    const float4 g0 = reinterpret_cast<const float4*>(gamma + lane * 8)[0];
    const float4 g1 = reinterpret_cast<const float4*>(gamma + lane * 8)[1];
    const float4 b0 = reinterpret_cast<const float4*>(beta + lane * 8)[0];
    const float4 b1 = reinterpret_cast<const float4*>(beta + lane * 8)[1];

    float4 o0, o1;
    o0.x = (v0.x - mean) * rstd * g0.x + b0.x;
    o0.y = (v0.y - mean) * rstd * g0.y + b0.y;
    o0.z = (v0.z - mean) * rstd * g0.z + b0.z;
    o0.w = (v0.w - mean) * rstd * g0.w + b0.w;
    o1.x = (v1.x - mean) * rstd * g1.x + b1.x;
    o1.y = (v1.y - mean) * rstd * g1.y + b1.y;
    o1.z = (v1.z - mean) * rstd * g1.z + b1.z;
    o1.w = (v1.w - mean) * rstd * g1.w + b1.w;

    reinterpret_cast<float4*>(p)[0] = o0;
    reinterpret_cast<float4*>(p)[1] = o1;
}

// ------------------------------- launcher ----------------------------------
extern "C" void kernel_launch(void* const* d_in, const int* in_sizes, int n_in,
                              void* d_out, int out_size, void* d_ws, size_t ws_size,
                              hipStream_t stream) {
    const float* x     = (const float*)d_in[0];
    const int*   src   = (const int*)d_in[1];
    const int*   dst   = (const int*)d_in[2];
    const float* Wself = (const float*)d_in[3];
    const float* Wrel  = (const float*)d_in[4];
    const float* gamma = (const float*)d_in[5];
    const float* beta  = (const float*)d_in[6];
    float* out = (float*)d_out;

    // workspace layout: x_bf16 [N*256] then Wt_bf16 [9*256*256]
    unsigned short* xb = (unsigned short*)d_ws;
    unsigned short* wb = xb + (size_t)N_NODES * DIM;

    // 1) fp32 -> bf16 conversions (x_bf16 = 51 MB, stays resident in 192 MB L2)
    k_convert_x<<<(N_NODES * DIM / 4 + 255) / 256, 256, 0, stream>>>(x, xb, N_NODES * DIM / 4);
    k_convert_w<<<(9 * DIM * DIM) / 256, 256, 0, stream>>>(Wself, Wrel, wb);

    // 2) self-loop GEMM (initializes every element of d_out)
    k_self_gemm<<<N_NODES / 16, 256, 0, stream>>>(xb, wb, out);

    // 3) per-relation gather-GEMM with LDS staging + atomic scatter-add
    k_edge_gemm<<<N_REL * (E_PER_REL / EDGE_TILE), 256, 0, stream>>>(xb, wb, src, dst, out);

    // 4) LayerNorm in place
    k_layernorm<<<N_NODES / 8, 256, 0, stream>>>(out, gamma, beta);
}
